// WindowAttention_49263274885582
// MI455X (gfx1250) — matrix operand order repaired and checked by reading, not compiled, not run on hardware
//
#include <hip/hip_runtime.h>
#include <hip/hip_bf16.h>

typedef __attribute__((ext_vector_type(16))) _Float16 v16h;
typedef __attribute__((ext_vector_type(8)))  float    v8f;

#define NTOK 49
#define NPAD 64
#define DIMC 256
#define QKVC 768
#define NHD  8
#define HDIM 32
#define ATT_SCALE 0.17677669529663687f   // 32^-0.5

__device__ __forceinline__ v8f wmma_f16(v16h a, v16h b, v8f c) {
  // D = A(16x32 f16) * B(32x16 f16) + C(16x16 f32)
  return __builtin_amdgcn_wmma_f32_16x16x32_f16(false, a, false, b, (short)0, c,
                                                false, false);
}

// Async bulk copy: global -> LDS, 16 bytes per lane, tracked by ASYNCcnt.
__device__ __forceinline__ void async_b128(void* lds_ptr, const void* gptr) {
  unsigned lds_addr = (unsigned)(unsigned long long)lds_ptr;  // LDS aperture low32
  asm volatile("global_load_async_to_lds_b128 %0, %1, off"
               :: "v"(lds_addr), "v"(gptr) : "memory");
}
__device__ __forceinline__ void wait_async0() {
  asm volatile("s_wait_asynccnt 0" ::: "memory");
}

// A-fragment (16x32 f16) from row-major src[m][k], ld = row stride in elements.
// Layout (wave32): lanes 0-15: M=lane, halves 0..7 -> K=0..7, 8..15 -> K=16..23
//                  lanes 16-31: same M, K runs shifted by +8.
__device__ __forceinline__ v16h load_frag_a(const _Float16* p0, int ld) {
  const int lane = threadIdx.x & 31;
  const _Float16* p = p0 + (lane & 15) * ld + ((lane >> 4) << 3);
  union { v16h v; float4 q[2]; } u;
  u.q[0] = *(const float4*)(p);
  u.q[1] = *(const float4*)(p + 16);
  return u.v;
}

// B-fragment (32x16 f16) from K-major src Bt[n][k], ld = row stride in elements.
// Layout (wave32): lane n = lane&15, halves 0..7 -> K=0..7, 8..15 -> K=16..23; lanes>=16 -> K+8.
__device__ __forceinline__ v16h load_frag_b(const _Float16* p0, int ld) {
  const int lane = threadIdx.x & 31;
  const _Float16* p = p0 + (lane & 15) * ld + ((lane >> 4) << 3);
  union { v16h v; float4 q[2]; } u;
  u.q[0] = *(const float4*)(p);
  u.q[1] = *(const float4*)(p + 16);
  return u.v;
}

// ---------------------------------------------------------------------------
// Prep: transpose+convert weights to f16 (K-major), expand relative-pos bias.
// ---------------------------------------------------------------------------
__global__ void prep_kernel(const float* __restrict__ qkv_w,   // [256][768]
                            const float* __restrict__ proj_w,  // [256][256]
                            const float* __restrict__ rpb,     // [169][8]
                            _Float16* __restrict__ qkvWt,      // [768][256]
                            _Float16* __restrict__ projWt,     // [256][256]
                            float* __restrict__ biasBuf) {     // [8][49][49]
  int i = blockIdx.x * 256 + threadIdx.x;
  if (i < QKVC * DIMC) {            // qkvWt[n][k] = qkv_w[k][n]
    int n = i >> 8, k = i & 255;
    qkvWt[i] = (_Float16)qkv_w[k * QKVC + n];
  }
  if (i < DIMC * DIMC) {            // projWt[n][k] = proj_w[k][n]
    int n = i >> 8, k = i & 255;
    projWt[i] = (_Float16)proj_w[k * DIMC + n];
  }
  if (i < NHD * NTOK * NTOK) {      // biasBuf[h][n][m]
    int h = i / (NTOK * NTOK);
    int rem = i % (NTOK * NTOK);
    int n = rem / NTOK, m = rem % NTOK;
    int iy = n / 7, ix = n % 7, jy = m / 7, jx = m % 7;
    int ridx = (iy - jy + 6) * 13 + (ix - jx + 6);
    biasBuf[i] = rpb[ridx * NHD + h];
  }
}

// ---------------------------------------------------------------------------
// Fused window attention: one window per block, 8 waves (one head per wave).
// ---------------------------------------------------------------------------
__global__ void __launch_bounds__(256, 1)
win_attn_kernel(const float* __restrict__ x,       // [4096][49][256]
                const float* __restrict__ mask,    // [4096][49][49]
                const float* __restrict__ qkv_b,   // [768]
                const float* __restrict__ proj_b,  // [256]
                const _Float16* __restrict__ qkvWt,
                const _Float16* __restrict__ projWt,
                const float* __restrict__ biasBuf,
                float* __restrict__ out) {         // [4096][49][256]
  __shared__ _Float16 qk[NPAD * 512];              // Q|K rows (also f32 x stage), 64 KB
  __shared__ _Float16 vT[NHD * HDIM * 72];         // V transposed (padded), 36 KB
  __shared__ _Float16 region1[NPAD * 512];         // xh | probs | Oall, 64 KB
  __shared__ float    masklds[NTOK * NTOK];        // 9.4 KB

  const int b    = blockIdx.x;
  const int tid  = threadIdx.x;
  const int lane = tid & 31;
  const int wave = tid >> 5;
  const int hi   = lane >> 4;     // half-wave select
  const int ln16 = lane & 15;

  // ---- async-stage x (f32, into qk region) and mask directly to LDS ----
  float* xstage = (float*)qk;                           // 49*256*4 = 50176 B
  const float* xb = x + (size_t)b * NTOK * DIMC;
  for (int c = tid; c < (NTOK * DIMC) / 4; c += 256)    // 3136 x 16B chunks
    async_b128(xstage + c * 4, xb + c * 4);
  const float* mb = mask + (size_t)b * NTOK * NTOK;
  for (int c = tid; c < (NTOK * NTOK) / 4; c += 256)    // 600 x 16B chunks
    async_b128(masklds + c * 4, mb + c * 4);
  if (tid == 0) masklds[2400] = mb[2400];               // tail element
  wait_async0();
  __syncthreads();

  // ---- convert x -> f16 xh (rows >= 49 zeroed) ----
  _Float16* xh = region1;                               // [64][256]
  for (int i4 = tid; i4 < (NPAD * DIMC) / 4; i4 += 256) {
    int row = (i4 << 2) >> 8;
    float4 v = make_float4(0.f, 0.f, 0.f, 0.f);
    if (row < NTOK) v = ((const float4*)xstage)[i4];
    _Float16* d = xh + (i4 << 2);
    d[0] = (_Float16)v.x; d[1] = (_Float16)v.y;
    d[2] = (_Float16)v.z; d[3] = (_Float16)v.w;
  }
  __syncthreads();

  // ---- qkv GEMM: [64x256] x [256x768]; B panel held in registers per nt ----
  for (int nt = wave; nt < 48; nt += 8) {
    v16h bfr[8];
#pragma unroll
    for (int k8 = 0; k8 < 8; k8++)
      bfr[k8] = load_frag_b(qkvWt + nt * 16 * DIMC + k8 * 32, DIMC);
    int n0  = nt * 16;
    int col = n0 + ln16;
    float bv = qkv_b[col];
    for (int mt = 0; mt < 4; mt++) {
      v8f acc = {};
#pragma unroll
      for (int k8 = 0; k8 < 8; k8++)
        acc = wmma_f16(load_frag_a(xh + mt * 16 * DIMC + k8 * 32, DIMC),
                       bfr[k8], acc);
      int rbase = mt * 16 + (hi << 3);
      if (n0 < 512) {                                // Q|K rows, row-major
#pragma unroll
        for (int r = 0; r < 8; r++)
          qk[(rbase + r) * 512 + col] = (_Float16)(acc[r] + bv);
      } else {                                       // V: store transposed
        int h = (col - 512) >> 5, d = (col - 512) & 31;
#pragma unroll
        for (int r = 0; r < 8; r++)
          vT[(h * HDIM + d) * 72 + (rbase + r)] = (_Float16)(acc[r] + bv);
      }
    }
  }
  __syncthreads();

  // ---- attention: one head per wave ----
  const int h = wave;
  _Float16* probs = region1;                          // [8][64][64] (xh is dead)
  const float* biasH = biasBuf + h * NTOK * NTOK;

  for (int mt = 0; mt < 4; mt++) {
    v8f s[4];
#pragma unroll
    for (int j = 0; j < 4; j++) {
      v16h a  = load_frag_a(qk + (mt * 16) * 512 + h * HDIM, 512);        // Q tile
      v16h bb = load_frag_b(qk + (j * 16) * 512 + 256 + h * HDIM, 512);   // K tile
      v8f z = {};
      s[j] = wmma_f16(a, bb, z);
    }
    // scale + bias + mask + padding mask (C layout: n = mt*16+hi*8+r, m = j*16+ln16)
#pragma unroll
    for (int j = 0; j < 4; j++) {
      int m = j * 16 + ln16;
#pragma unroll
      for (int r = 0; r < 8; r++) {
        int n = mt * 16 + (hi << 3) + r;
        float v = -1e30f;
        if (m < NTOK && n < NTOK)
          v = s[j][r] * ATT_SCALE + biasH[n * NTOK + m] + masklds[n * NTOK + m];
        s[j][r] = v;
      }
    }
    // row softmax: reduce over j lanewise, then across the 16-lane half.
#pragma unroll
    for (int r = 0; r < 8; r++) {
      float mx = fmaxf(fmaxf(s[0][r], s[1][r]), fmaxf(s[2][r], s[3][r]));
#pragma unroll
      for (int off = 1; off < 16; off <<= 1)
        mx = fmaxf(mx, __shfl_xor(mx, off, 32));
      float e0 = __expf(s[0][r] - mx), e1 = __expf(s[1][r] - mx);
      float e2 = __expf(s[2][r] - mx), e3 = __expf(s[3][r] - mx);
      float sm = e0 + e1 + e2 + e3;
#pragma unroll
      for (int off = 1; off < 16; off <<= 1)
        sm += __shfl_xor(sm, off, 32);
      float inv = __frcp_rn(sm);
      int n = mt * 16 + (hi << 3) + r;
      _Float16* prow = probs + h * 4096 + n * 64 + ln16;
      prow[0]  = (_Float16)(e0 * inv);
      prow[16] = (_Float16)(e1 * inv);
      prow[32] = (_Float16)(e2 * inv);
      prow[48] = (_Float16)(e3 * inv);
    }
  }

  // ---- P @ V : [64x64] x [64x32] per head ----
  v8f o[4][2];
  for (int mt = 0; mt < 4; mt++) {
#pragma unroll
    for (int nt = 0; nt < 2; nt++) {
      v8f acc = {};
#pragma unroll
      for (int k = 0; k < 64; k += 32) {
        v16h a  = load_frag_a(probs + h * 4096 + (mt * 16) * 64 + k, 64);
        v16h bb = load_frag_b(vT + h * HDIM * 72 + (nt * 16) * 72 + k, 72);
        acc = wmma_f16(a, bb, acc);
      }
      o[mt][nt] = acc;
    }
  }
  __syncthreads();                       // all probs consumed before Oall overwrite

  _Float16* Oall = region1;              // [64][256]
  for (int mt = 0; mt < 4; mt++) {
#pragma unroll
    for (int nt = 0; nt < 2; nt++) {
      int col   = h * HDIM + nt * 16 + ln16;
      int rbase = mt * 16 + (hi << 3);
#pragma unroll
      for (int r = 0; r < 8; r++)
        Oall[(rbase + r) * DIMC + col] = (_Float16)o[mt][nt][r];
    }
  }
  __syncthreads();

  // ---- proj GEMM: [64x256] x [256x256]; B panel held in registers per nt ----
  for (int ntp = 0; ntp < 2; ntp++) {
    int nt = wave * 2 + ntp;
    v16h bfr[8];
#pragma unroll
    for (int k8 = 0; k8 < 8; k8++)
      bfr[k8] = load_frag_b(projWt + nt * 16 * DIMC + k8 * 32, DIMC);
    int n0  = nt * 16;
    int col = n0 + ln16;
    float pb = proj_b[col];
    float* ob = out + (size_t)b * NTOK * DIMC;
    for (int mt = 0; mt < 4; mt++) {
      v8f acc = {};
#pragma unroll
      for (int k8 = 0; k8 < 8; k8++)
        acc = wmma_f16(load_frag_a(Oall + mt * 16 * DIMC + k8 * 32, DIMC),
                       bfr[k8], acc);
      int rbase = mt * 16 + (hi << 3);
#pragma unroll
      for (int r = 0; r < 8; r++) {
        int row = rbase + r;
        if (row < NTOK) ob[row * DIMC + col] = acc[r] + pb;
      }
    }
  }
}

extern "C" void kernel_launch(void* const* d_in, const int* in_sizes, int n_in,
                              void* d_out, int out_size, void* d_ws, size_t ws_size,
                              hipStream_t stream) {
  const float* x      = (const float*)d_in[0];
  const float* mask   = (const float*)d_in[1];
  const float* qkv_w  = (const float*)d_in[2];
  const float* qkv_b  = (const float*)d_in[3];
  const float* rpb    = (const float*)d_in[4];
  const float* proj_w = (const float*)d_in[5];
  const float* proj_b = (const float*)d_in[6];
  float* out = (float*)d_out;

  char* ws = (char*)d_ws;
  _Float16* qkvWt  = (_Float16*)ws;                          // 768*256*2 = 393216
  _Float16* projWt = (_Float16*)(ws + 393216);               // 256*256*2 = 131072
  float*    biasBuf = (float*)(ws + 393216 + 131072);        // 8*49*49*4 = 76832

  prep_kernel<<<(QKVC * DIMC + 255) / 256, 256, 0, stream>>>(
      qkv_w, proj_w, rpb, qkvWt, projWt, biasBuf);

  win_attn_kernel<<<4096, 256, 0, stream>>>(
      x, mask, qkv_b, proj_b, qkvWt, projWt, biasBuf, out);
}
